// ImageBppEstimator_50328426775006
// MI455X (gfx1250) — matrix operand order repaired and checked by reading, not compiled, not run on hardware
//
#include <hip/hip_runtime.h>
#include <cstddef>
#include <cstdint>

// ===================== CDNA5 WMMA types & helpers =====================
typedef __bf16 bf16;
typedef __attribute__((ext_vector_type(16))) __bf16 v16bf;
typedef __attribute__((ext_vector_type(2)))  __bf16 v2bf;
typedef __attribute__((ext_vector_type(8)))  float  v8f;
typedef __attribute__((ext_vector_type(4)))  float  f4v;
typedef __attribute__((ext_vector_type(4)))  unsigned int uint4v;

// hardware f32->bf16 (ISel should pick v_cvt_pk_bf16_f32 for pairs)
__device__ static inline unsigned short f2bf(float f) {
  return __builtin_bit_cast(unsigned short, (__bf16)f);
}
__device__ static inline unsigned pkbf(float a, float b) {
  v2bf h; h[0] = (__bf16)a; h[1] = (__bf16)b;
  return __builtin_bit_cast(unsigned, h);
}
// pack 8 bf16 into one 16B dword4 for a single ds_store_b128
__device__ static inline uint4v pack8(float a0, float a1, float a2, float a3,
                                      float a4, float a5, float a6, float a7) {
  uint4v p;
  p.x = pkbf(a0, a1); p.y = pkbf(a2, a3); p.z = pkbf(a4, a5); p.w = pkbf(a6, a7);
  return p;
}
// A-fragment row permutation: group g of 8 consecutive k maps to row offset
//   g0(k0..7)->0, g1(k8..15)->16, g2(k16..23)->8, g3(k24..31)->24
// so lane-half 0 reads [k0..7,k16..23], half 1 reads [k8..15,k24..31] contiguously.
__device__ static inline int agofs(int g) { return ((g & 1) << 4) + ((g >> 1) << 3); }

// ===================== generic bf16 WMMA GEMM =====================
// out = epilogue(A[M,K] @ W[K,N] + bias), 128x64 block tile, 128 threads (4 waves),
// each wave computes a 32x64 stripe (8 WMMA accumulators -> 0.75 frag loads/WMMA).
// act: 0 none, 1 gelu(exact), 2 relu, 3 sigmoid. mod: per-(batch,col) scale.
__global__ __launch_bounds__(128) void gemm_bf16_wmma(
    const float* __restrict__ A, const float* __restrict__ Wt,
    const float* __restrict__ bias, const float* __restrict__ res,
    const float* __restrict__ mod, float* __restrict__ out,
    int M, int N, int K, int act, int rowsPerBatch)
{
  __shared__ v16bf AsV[128 * 3];  // 128 rows, stride 48 ushort (A-frag order, padded)
  __shared__ v16bf BsV[64 * 3];   // 64 cols, stride 48 ushort (N-major, K contiguous)
  unsigned short* Asu = (unsigned short*)AsV;
  unsigned short* Bsu = (unsigned short*)BsV;

  int tid  = threadIdx.x;
  int row0 = blockIdx.y << 7;
  int col0 = blockIdx.x << 6;
  int wave = tid >> 5, lane = tid & 31;
  int l15  = lane & 15, hi = lane >> 4;

  v8f zero = {};
  v8f acc0[4] = {zero, zero, zero, zero};
  v8f acc1[4] = {zero, zero, zero, zero};

  for (int k0 = 0; k0 < K; k0 += 32) {
    // ---- stage A: 128 rows x 32 k, chunk = (row, group-of-8-k), b128 loads/stores ----
    for (int c = tid; c < 512; c += 128) {
      int r = c >> 2, g = c & 3;
      int gr = row0 + r;
      f4v v01 = {}, v23 = {};
      if (gr < M) {
        const f4v* pa = (const f4v*)(A + (size_t)gr * K + k0 + (g << 3));
        v01 = pa[0]; v23 = pa[1];
      }
      *(uint4v*)(Asu + r * 48 + agofs(g)) =
          pack8(v01.x, v01.y, v01.z, v01.w, v23.x, v23.y, v23.z, v23.w);
    }
    // ---- stage B with register transpose: chunk = (col, group-of-8-k) ----
    for (int c = tid; c < 256; c += 128) {
      int n = c & 63, g = c >> 6;
      int gn = col0 + n;
      float t0 = 0.f, t1 = 0.f, t2 = 0.f, t3 = 0.f, t4 = 0.f, t5 = 0.f, t6 = 0.f, t7 = 0.f;
      if (gn < N) {
        const float* pb = Wt + (size_t)(k0 + (g << 3)) * N + gn;
        t0 = pb[0];          t1 = pb[(size_t)N];     t2 = pb[(size_t)2 * N]; t3 = pb[(size_t)3 * N];
        t4 = pb[(size_t)4 * N]; t5 = pb[(size_t)5 * N]; t6 = pb[(size_t)6 * N]; t7 = pb[(size_t)7 * N];
      }
      *(uint4v*)(Bsu + n * 48 + agofs(g)) = pack8(t0, t1, t2, t3, t4, t5, t6, t7);
    }
    if (k0 + 32 < K)  // global_prefetch_b8 for next A tile
      __builtin_prefetch(A + (size_t)(row0 + tid) * K + k0 + 32, 0, 0);
    __syncthreads();

    v16bf a0 = AsV[((wave << 5) + l15) * 3 + hi];        // 2x ds_load_b128
    v16bf a1 = AsV[((wave << 5) + 16 + l15) * 3 + hi];   // 2x ds_load_b128
#pragma unroll
    for (int ni = 0; ni < 4; ++ni) {
      v16bf b = BsV[((ni << 4) + l15) * 3 + hi];         // 2x ds_load_b128
      acc0[ni] = __builtin_amdgcn_wmma_f32_16x16x32_bf16(false, a0, false, b,
                                                         (short)0, acc0[ni], false, false);
      acc1[ni] = __builtin_amdgcn_wmma_f32_16x16x32_bf16(false, a1, false, b,
                                                         (short)0, acc1[ni], false, false);
    }
    __syncthreads();
  }

#pragma unroll
  for (int mi = 0; mi < 2; ++mi) {
    int mbase = (wave << 5) + (mi << 4) + (hi << 3);
#pragma unroll
    for (int ni = 0; ni < 4; ++ni) {
      int col = col0 + (ni << 4) + l15;
      if (col >= N) continue;
      float bv = bias ? bias[col] : 0.f;
#pragma unroll
      for (int v = 0; v < 8; ++v) {
        int row = row0 + mbase + v;
        if (row >= M) continue;
        float x = (mi ? acc1[ni][v] : acc0[ni][v]) + bv;
        if (act == 1)      x = 0.5f * x * (1.f + erff(x * 0.70710678118654752f));
        else if (act == 2) x = x > 0.f ? x : 0.f;
        else if (act == 3) x = 1.f / (1.f + __expf(-x));
        if (mod) x *= mod[(size_t)(row / rowsPerBatch) * N + col];
        if (res) x += res[(size_t)row * N + col];
        out[(size_t)row * N + col] = x;
      }
    }
  }
}

// ===================== wave32 LayerNorm (1 wave / row) =====================
__global__ __launch_bounds__(256) void layernorm_k(
    const float* __restrict__ in, float* __restrict__ out,
    const float* __restrict__ g, const float* __restrict__ b, int M, int C)
{
  int wave = threadIdx.x >> 5, lane = threadIdx.x & 31;
  int row = (blockIdx.x << 3) + wave;
  if (row >= M) return;
  const float* x = in + (size_t)row * C;
  float s = 0.f, s2 = 0.f;
  for (int c = lane; c < C; c += 32) { float v = x[c]; s += v; s2 += v * v; }
  for (int m = 16; m > 0; m >>= 1) {
    s  += __shfl_xor(s,  m, 32);
    s2 += __shfl_xor(s2, m, 32);
  }
  float mean = s / (float)C;
  float var  = s2 / (float)C - mean * mean;
  float r    = rsqrtf(var + 1e-5f);
  float* o = out + (size_t)row * C;
  for (int c = lane; c < C; c += 32) o[c] = (x[c] - mean) * r * g[c] + b[c];
}

// ===================== patch embed: 2x2 s2 conv -> [B,L,128] =====================
__global__ void patch_embed_k(const float* __restrict__ x, const float* __restrict__ w,
                              const float* __restrict__ b, float* __restrict__ out, int B)
{
  size_t gid = (size_t)blockIdx.x * blockDim.x + threadIdx.x;
  size_t total = (size_t)B * 16384 * 128;
  if (gid >= total) return;
  int c  = (int)(gid % 128);
  size_t t = gid / 128;
  int p  = (int)(t % 16384);
  int bb = (int)(t / 16384);
  int y = p >> 7, xx = p & 127;
  float acc = b[c];
#pragma unroll
  for (int ci = 0; ci < 3; ++ci)
#pragma unroll
    for (int ky = 0; ky < 2; ++ky)
#pragma unroll
      for (int kx = 0; kx < 2; ++kx)
        acc += x[(size_t)(bb * 3 + ci) * 65536 + (size_t)(2 * y + ky) * 256 + (2 * xx + kx)]
             * w[((c * 3 + ci) * 2 + ky) * 2 + kx];
  out[gid] = acc;
}

// ===================== patch merge gather: [B,(2H)^2,C] -> [B,H^2,4C] =====================
__global__ void merge_gather_k(const float* __restrict__ in, float* __restrict__ out,
                               int B, int Hout, int C)
{
  size_t gid = (size_t)blockIdx.x * blockDim.x + threadIdx.x;
  size_t total = (size_t)B * Hout * Hout * 4 * C;
  if (gid >= total) return;
  int fourC = 4 * C;
  int j = (int)(gid % fourC);
  size_t t = gid / fourC;
  int p = (int)(t % (Hout * Hout));
  int b = (int)(t / (Hout * Hout));
  int part = j / C, c = j % C;
  int dy = part & 1, dx = part >> 1;          // parts: (0,0),(1,0),(0,1),(1,1)
  int y = p / Hout, xx = p % Hout;
  int Hin = 2 * Hout;
  out[gid] = in[((size_t)b * Hin * Hin + (size_t)(2 * y + dy) * Hin + (2 * xx + dx)) * C + c];
}

// ===================== window attention (1 block = 1 window*head), d=32, N=64 =====================
__global__ __launch_bounds__(128) void win_attn_k(
    const float* __restrict__ qkv, const float* __restrict__ rpb,
    float* __restrict__ outp, int B, int Himg, int C, int nheads, int shift)
{
  __shared__ v16bf QsV[64 * 3];          // A-frag order, stride 48 ushort
  __shared__ v16bf KsV[64 * 3];          // token-major (= B N-major), stride 48 ushort
  __shared__ v16bf VtV[32 * 5];          // V^T [dim][token], stride 80 ushort
  __shared__ v16bf PbV[64 * 5];          // probs, A-frag order, stride 80 ushort
  __shared__ float Pf[64 * 65];          // scores, padded stride 65
  __shared__ unsigned long long tokOff[64];
  unsigned short* Qsu = (unsigned short*)QsV;
  unsigned short* Ksu = (unsigned short*)KsV;
  unsigned short* Vtu = (unsigned short*)VtV;
  unsigned short* Pbu = (unsigned short*)PbV;

  int nwx  = Himg >> 3;
  int nW   = nwx * nwx;
  int id   = blockIdx.x;
  int head = id % nheads;
  int win  = id / nheads;
  int b    = win / nW;
  int wr_  = win % nW;
  int wy = wr_ / nwx, wx = wr_ % nwx;

  int tid = threadIdx.x;
  const float scale = 0.17677669529663687f;   // 1/sqrt(32)
  const int C3 = 3 * C;

  if (tid < 64) {                       // roll(-shift) token gather table
    int hr = (wy << 3) + (tid >> 3);
    int wc = (wx << 3) + (tid & 7);
    int h = hr + shift; if (h >= Himg) h -= Himg;
    int w = wc + shift; if (w >= Himg) w -= Himg;
    tokOff[tid] = (unsigned long long)b * Himg * Himg + (unsigned long long)h * Himg + w;
  }
  __syncthreads();

  // ---- stage Q (A-frag order, pre-scaled) and K (N-major): b128 loads + b128 stores ----
  for (int c = tid; c < 256; c += 128) {
    int t = c >> 2, g = c & 3;
    const float* p = qkv + (size_t)tokOff[t] * C3 + head * 32 + (g << 3);
    f4v q01 = ((const f4v*)p)[0], q23 = ((const f4v*)p)[1];
    *(uint4v*)(Qsu + t * 48 + agofs(g)) =
        pack8(q01.x * scale, q01.y * scale, q01.z * scale, q01.w * scale,
              q23.x * scale, q23.y * scale, q23.z * scale, q23.w * scale);
    const f4v* pk = (const f4v*)(p + C);
    f4v k01 = pk[0], k23 = pk[1];
    *(uint4v*)(Ksu + t * 48 + agofs(g)) =
        pack8(k01.x, k01.y, k01.z, k01.w, k23.x, k23.y, k23.z, k23.w);
  }
  // ---- stage V transposed: chunk = (dim, group-of-8-tokens) ----
  for (int c = tid; c < 256; c += 128) {
    int d = c & 31, tg = c >> 5;
    float t8[8];
#pragma unroll
    for (int j = 0; j < 8; ++j) {
      int t = (tg << 3) + j;
      t8[j] = qkv[(size_t)tokOff[t] * C3 + 2 * C + head * 32 + d];
    }
    *(uint4v*)(Vtu + d * 80 + ((tg & 4) << 3) + agofs(tg & 3)) =
        pack8(t8[0], t8[1], t8[2], t8[3], t8[4], t8[5], t8[6], t8[7]);
  }
  __syncthreads();

  int wave = tid >> 5, lane = tid & 31;
  int l15 = lane & 15, hi = lane >> 4;
  int mbase = (wave << 4) + (hi << 3);

  {   // S = Q K^T (+ relative-position bias + shift mask), per-wave 16-row stripe
    v16bf a = QsV[((wave << 4) + l15) * 3 + hi];
    v8f zero = {};
#pragma unroll
    for (int ni = 0; ni < 4; ++ni) {
      v16bf bfr = KsV[((ni << 4) + l15) * 3 + hi];
      v8f acc = __builtin_amdgcn_wmma_f32_16x16x32_bf16(false, a, false, bfr,
                                                        (short)0, zero, false, false);
      int kc = (ni << 4) + l15;                  // key token
      int yk = kc >> 3, xk = kc & 7;
#pragma unroll
      for (int v = 0; v < 8; ++v) {
        int qc = mbase + v;                      // query token
        int yq = qc >> 3, xq = qc & 7;
        int ridx = (yq - yk + 7) * 15 + (xq - xk + 7);
        float biasv = rpb[ridx * nheads + head];
        float maskv = 0.f;
        if (shift > 0) {
          int hq = (wy << 3) + yq, wq = (wx << 3) + xq;
          int hk = (wy << 3) + yk, wk = (wx << 3) + xk;
          int lq = ((hq < Himg - 8) ? 0 : (hq < Himg - shift) ? 1 : 2) * 3 +
                   ((wq < Himg - 8) ? 0 : (wq < Himg - shift) ? 1 : 2);
          int lk = ((hk < Himg - 8) ? 0 : (hk < Himg - shift) ? 1 : 2) * 3 +
                   ((wk < Himg - 8) ? 0 : (wk < Himg - shift) ? 1 : 2);
          if (lq != lk) maskv = -100.f;
        }
        Pf[qc * 65 + kc] = acc[v] + biasv + maskv;
      }
    }
  }
  __syncthreads();

  if (tid < 64) {                  // softmax; write probs straight into A-frag order
    float mx = -1e30f;
    for (int c = 0; c < 64; ++c) mx = fmaxf(mx, Pf[tid * 65 + c]);
    float sum = 0.f;
    for (int c = 0; c < 64; ++c) { float e = __expf(Pf[tid * 65 + c] - mx); Pf[tid * 65 + c] = e; sum += e; }
    float inv = 1.f / sum;
    for (int c = 0; c < 64; ++c) {
      int k = c & 31;
      int slot = (k < 8) ? k : (k < 16) ? (k + 8) : (k < 24) ? (k - 8) : k;
      Pbu[tid * 80 + (c & 32) + slot] = f2bf(Pf[tid * 65 + c] * inv);
    }
  }
  __syncthreads();

#pragma unroll
  for (int ni = 0; ni < 2; ++ni) {               // O = P V
    v8f acc = {};
#pragma unroll
    for (int kb = 0; kb < 2; ++kb) {
      v16bf a   = PbV[((wave << 4) + l15) * 5 + (kb << 1) + hi];
      v16bf bfr = VtV[((ni << 4) + l15) * 5 + (kb << 1) + hi];
      acc = __builtin_amdgcn_wmma_f32_16x16x32_bf16(false, a, false, bfr,
                                                    (short)0, acc, false, false);
    }
    int dn = (ni << 4) + l15;
#pragma unroll
    for (int v = 0; v < 8; ++v) {
      int qc = mbase + v;
      outp[(size_t)tokOff[qc] * C + head * 32 + dn] = acc[v];
    }
  }
}

// ===================== tiny per-batch modulation MLP (1->480->480->480 sigmoid) ==========
__global__ __launch_bounds__(512) void modnet_k(
    const float* __restrict__ QP,
    const float* __restrict__ w1, const float* __restrict__ b1,
    const float* __restrict__ w2, const float* __restrict__ b2,
    const float* __restrict__ w3, const float* __restrict__ b3,
    float* __restrict__ mout)
{
  const int HID = 480;
  __shared__ float h1[480];
  __shared__ float h2[480];
  int b = blockIdx.x, j = threadIdx.x;
  float q = QP[b];
  if (j < HID) { float v = q * w1[j] + b1[j]; h1[j] = v > 0.f ? v : 0.f; }
  __syncthreads();
  if (j < HID) {
    float a = b2[j];
    for (int k = 0; k < HID; ++k) a += h1[k] * w2[k * HID + j];
    h2[j] = a > 0.f ? a : 0.f;
  }
  __syncthreads();
  if (j < HID) {
    float a = b3[j];
    for (int k = 0; k < HID; ++k) a += h2[k] * w3[k * HID + j];
    mout[b * HID + j] = 1.f / (1.f + __expf(-a));
  }
}

__global__ void elem_mul_k(const float* __restrict__ a, const float* __restrict__ b,
                           float* __restrict__ o, size_t n)
{
  size_t i = (size_t)blockIdx.x * blockDim.x + threadIdx.x;
  if (i < n) o[i] = a[i] * b[i];
}

__global__ __launch_bounds__(256) void head_dot_k(
    const float* __restrict__ h, const float* __restrict__ w,
    const float* __restrict__ bb, float* __restrict__ out, int per)
{
  __shared__ float red[256];
  int b = blockIdx.x;
  const float* x = h + (size_t)b * per;
  float s = 0.f;
  for (int i = threadIdx.x; i < per; i += 256) s += x[i] * w[i];
  red[threadIdx.x] = s;
  __syncthreads();
  for (int st = 128; st > 0; st >>= 1) {
    if ((int)threadIdx.x < st) red[threadIdx.x] += red[threadIdx.x + st];
    __syncthreads();
  }
  if (threadIdx.x == 0) out[b] = red[0] + bb[0];
}

// ===================== host side =====================
static inline void launch_gemm(const float* A, const float* W, const float* bias,
                               const float* res, const float* mod, float* out,
                               int M, int N, int K, int act, int rpb, hipStream_t s)
{
  dim3 g((N + 63) / 64, (M + 127) / 128);
  gemm_bf16_wmma<<<g, 128, 0, s>>>(A, W, bias, res, mod, out, M, N, K, act, rpb);
}

extern "C" void kernel_launch(void* const* d_in, const int* in_sizes, int n_in,
                              void* d_out, int out_size, void* d_ws, size_t ws_size,
                              hipStream_t stream)
{
  (void)in_sizes; (void)n_in; (void)out_size; (void)ws_size;

  const float* x  = (const float*)d_in[0];
  const float* QP = (const float*)d_in[1];

  // ---- unpack params (jax pytree order: dict keys sorted alphabetically) ----
  int cur = 2;
  auto NXT = [&]() { return (const float*)d_in[cur++]; };

  const float* head_b = NXT();
  const float* head_w = NXT();

  struct BlkP { const float *fc1b,*fc1w,*fc2b,*fc2w,*n1b,*n1g,*n2b,*n2g,
                            *projb,*projw,*qkvb,*qkvw,*rpb; };
  struct MrgP { const float *nb,*ng,*red; };
  const int DP[4] = {2, 2, 6, 2};
  const int blkofs[4] = {0, 2, 4, 10};
  BlkP blocks[12];
  MrgP merges[4];
  for (int i = 0; i < 4; ++i) {
    for (int d = 0; d < DP[i]; ++d) {
      BlkP& K = blocks[blkofs[i] + d];
      K.fc1b = NXT(); K.fc1w = NXT(); K.fc2b = NXT(); K.fc2w = NXT();
      K.n1b = NXT();  K.n1g = NXT();  K.n2b = NXT();  K.n2g = NXT();
      K.projb = NXT(); K.projw = NXT(); K.qkvb = NXT(); K.qkvw = NXT();
      K.rpb = NXT();
    }
    if (i > 0) { merges[i].nb = NXT(); merges[i].ng = NXT(); merges[i].red = NXT(); }
  }
  struct ModNet {
    const float *bmb1[7], *bmw1[7], *bmb2[7], *bmw2[7], *bmb3[7], *bmw3[7];
    const float *smb[8], *smw[8];
  };
  ModNet mods[2];
  for (int m = 0; m < 2; ++m) {
    for (int i = 0; i < 7; ++i) {
      mods[m].bmb1[i] = NXT(); mods[m].bmw1[i] = NXT();
      mods[m].bmb2[i] = NXT(); mods[m].bmw2[i] = NXT();
      mods[m].bmb3[i] = NXT(); mods[m].bmw3[i] = NXT();
    }
    for (int i = 0; i < 8; ++i) { mods[m].smb[i] = NXT(); mods[m].smw[i] = NXT(); }
  }
  const float* fin_nb = NXT();
  const float* fin_ng = NXT();
  const float* pe_b  = NXT();
  const float* pe_nb = NXT();
  const float* pe_ng = NXT();
  const float* pe_w  = NXT();

  // ---- workspace layout ----
  char* ws = (char*)d_ws;
  const size_t SZ = 67108864ull;                    // 64 MB = 8*16384*128 floats
  float* bufA = (float*)(ws);
  float* bufB = (float*)(ws + SZ);
  float* hN   = (float*)(ws + 2 * SZ);
  float* attO = (float*)(ws + 3 * SZ);
  float* big  = (float*)(ws + 4 * SZ);              // 256 MB: qkv / mlp / val (time-shared)
  float* msc  = (float*)(ws + 4 * SZ + 268435456ull);

  const int Bn = 8;
  const int Hs[4] = {128, 64, 32, 16};
  const int Cs[4] = {128, 192, 256, 320};
  const int NH[4] = {4, 6, 8, 10};

  // ---- patch embed + LN ----
  {
    size_t tot = (size_t)Bn * 16384 * 128;
    patch_embed_k<<<(unsigned)((tot + 255) / 256), 256, 0, stream>>>(x, pe_w, pe_b, bufB, Bn);
    int M = Bn * 16384;
    layernorm_k<<<(M + 7) / 8, 256, 0, stream>>>(bufB, bufA, pe_ng, pe_nb, M, 128);
  }

  float* xc = bufA;
  float* xo = bufB;
  int bi = 0;
  for (int s = 0; s < 4; ++s) {
    int H = Hs[s], C = Cs[s], L = H * H, M = Bn * L;
    if (s > 0) {
      int Cp = Cs[s - 1];
      size_t tot = (size_t)M * 4 * Cp;
      merge_gather_k<<<(unsigned)((tot + 255) / 256), 256, 0, stream>>>(xc, attO, Bn, H, Cp);
      layernorm_k<<<(M + 7) / 8, 256, 0, stream>>>(attO, hN, merges[s].ng, merges[s].nb, M, 4 * Cp);
      launch_gemm(hN, merges[s].red, nullptr, nullptr, nullptr, xo, M, C, 4 * Cp, 0, 1, stream);
      { float* t = xc; xc = xo; xo = t; }
    }
    int nW = (H / 8) * (H / 8);
    for (int d = 0; d < DP[s]; ++d, ++bi) {
      const BlkP& K = blocks[bi];
      int shift = (d & 1) ? 4 : 0;
      layernorm_k<<<(M + 7) / 8, 256, 0, stream>>>(xc, hN, K.n1g, K.n1b, M, C);
      launch_gemm(hN, K.qkvw, K.qkvb, nullptr, nullptr, big, M, 3 * C, C, 0, 1, stream);
      win_attn_k<<<Bn * nW * NH[s], 128, 0, stream>>>(big, K.rpb, attO, Bn, H, C, NH[s], shift);
      launch_gemm(attO, K.projw, K.projb, xc, nullptr, xo, M, C, C, 0, 1, stream);
      { float* t = xc; xc = xo; xo = t; }
      layernorm_k<<<(M + 7) / 8, 256, 0, stream>>>(xc, hN, K.n2g, K.n2b, M, C);
      launch_gemm(hN, K.fc1w, K.fc1b, nullptr, nullptr, big, M, 4 * C, C, 1, 1, stream);
      launch_gemm(big, K.fc2w, K.fc2b, xc, nullptr, xo, M, C, 4 * C, 0, 1, stream);
      { float* t = xc; xc = xo; xo = t; }
    }
  }

  // ---- final LN ----
  int M = Bn * 256;
  layernorm_k<<<(M + 7) / 8, 256, 0, stream>>>(xc, xo, fin_ng, fin_nb, M, 320);
  { float* t = xc; xc = xo; xo = t; }

  // ---- two modulation networks ----
  for (int mi = 0; mi < 2; ++mi) {
    const ModNet& Mo = mods[mi];
    float* tin = hN;
    float* tout = attO;
    for (int i = 0; i < 7; ++i) {
      modnet_k<<<Bn, 512, 0, stream>>>(QP, Mo.bmw1[i], Mo.bmb1[i], Mo.bmw2[i], Mo.bmb2[i],
                                       Mo.bmw3[i], Mo.bmb3[i], msc);
      const float* src = (i == 0) ? xc : tin;
      int K = (i == 0) ? 320 : 480;
      launch_gemm(src, Mo.smw[i], Mo.smb[i], nullptr, msc, tout, M, 480, K, 0, 256, stream);
      { float* t = tin; tin = tout; tout = t; }
    }
    launch_gemm(tin, Mo.smw[7], Mo.smb[7], nullptr, nullptr, big, M, 320, 480, 3, 1, stream);
    size_t tot = (size_t)M * 320;
    elem_mul_k<<<(unsigned)((tot + 255) / 256), 256, 0, stream>>>(xc, big, xo, tot);
    { float* t = xc; xc = xo; xo = t; }
  }

  // ---- head ----
  head_dot_k<<<Bn, 256, 0, stream>>>(xc, head_w, head_b, (float*)d_out, 256 * 320);
}
